// MultiCNN_65395172049281
// MI455X (gfx1250) — matrix operand order repaired and checked by reading, not compiled, not run on hardware
//
#include <hip/hip_runtime.h>
#include <hip/hip_bf16.h>

typedef __attribute__((ext_vector_type(16))) _Float16 v16h;
typedef __attribute__((ext_vector_type(8)))  float    v8f;
typedef __attribute__((ext_vector_type(4)))  int      i32x4;

#define B_      8
#define N_      2500
#define NP_     2528      // N padded to multiple of 32 (79 chunks)
#define E_      100
#define F_      128
#define D_      384       // 3*F
#define Y_      8921
#define YT_     140       // ceil(Y/64): 64 y-rows per workgroup (16 per wave)
#define KSTEPS  12        // D/32
#define NCHUNK  79        // NP_/32
#define HST     400       // LDS h-chunk row stride in halfs (padded vs 64 banks)

#define AS1 __attribute__((address_space(1)))
#define AS3 __attribute__((address_space(3)))

// Async global->LDS path (ASYNCcnt) if this toolchain exposes the builtins.
#if defined(__has_builtin)
#  if __has_builtin(__builtin_amdgcn_global_load_async_to_lds_b128) && \
      __has_builtin(__builtin_amdgcn_s_wait_asynccnt)
#    define USE_ASYNC_LDS 1
#  endif
#endif

// ---------------------------------------------------------------------------
// Kernel 1: embedding gather + 3x conv1d('same') + tanh  ->  h[b][n][d] (f16)
// d = kidx*128 + f, kidx in {0,1,2} for k in {3,5,7}. Padded rows n>=N_ -> 0.
// ---------------------------------------------------------------------------
__global__ void conv_embed_tanh(const int* __restrict__ x,
                                const float* __restrict__ Wemb,
                                const float* __restrict__ w3, const float* __restrict__ b3,
                                const float* __restrict__ w5, const float* __restrict__ b5,
                                const float* __restrict__ w7, const float* __restrict__ b7,
                                _Float16* __restrict__ h) {
    long idx = (long)blockIdx.x * blockDim.x + threadIdx.x;
    const long total = (long)B_ * NP_ * D_;
    if (idx >= total) return;
    int d = (int)(idx % D_);
    int n = (int)((idx / D_) % NP_);
    int b = (int)(idx / ((long)D_ * NP_));
    if (n >= N_) { h[idx] = (_Float16)0.f; return; }
    int kidx = d / F_, f = d % F_;
    int k = (kidx == 0) ? 3 : ((kidx == 1) ? 5 : 7);
    const float* w  = (kidx == 0) ? w3 : ((kidx == 1) ? w5 : w7);
    const float* bb = (kidx == 0) ? b3 : ((kidx == 1) ? b5 : b7);
    float acc = bb[f];
    int pad = k >> 1;
    for (int j = 0; j < k; ++j) {
        int nn = n + j - pad;
        if (nn < 0 || nn >= N_) continue;                 // 'same' zero padding
        const float* eb = Wemb + (long)x[b * N_ + nn] * E_;
        const float* wb = w + (long)f * E_ * k + j;       // w[f][i][j], stride k over i
        #pragma unroll 4
        for (int i = 0; i < E_; ++i) acc = fmaf(wb[i * k], eb[i], acc);
    }
    h[idx] = (_Float16)tanhf(acc);
}

// ---------------------------------------------------------------------------
// WMMA fragment builders (wave32; A/B use this file's contiguous k order, not the ISA table order)
// A 16x32 f16: lane L holds row M=L&15; VGPR v holds K = 2v
//              + (L>=16 ? 16 : 0), two consecutive K per VGPR.
// B 32x16 f16: lane L holds col N=L&15; VGPR v holds K = 2v + (L>=16 ? 16 : 0).
// C/D 16x16 f32: lane L holds col N=L&15, VGPR v holds row M = v + 8*(L>=16).
// ---------------------------------------------------------------------------
__device__ inline v16h load_a_frag(const float* __restrict__ W, int y0,
                                   int lane, int kstep, int yMax) {
    int m  = lane & 15;
    int kb = (lane >> 4) * 16;
    int y  = y0 + m;
    v16h a;
    if (y < yMax) {
        const float* row = W + (long)y * D_ + kstep * 32;
        #pragma unroll
        for (int v = 0; v < 8; ++v) {
            int k0 = 2 * v + kb;
            a[2 * v]     = (_Float16)row[k0];
            a[2 * v + 1] = (_Float16)row[k0 + 1];
        }
    } else {
        #pragma unroll
        for (int i = 0; i < 16; ++i) a[i] = (_Float16)0.f;
    }
    return a;
}

__device__ inline v16h load_b_frag(const _Float16* __restrict__ hs, int tile,
                                   int lane, int kstep) {
    int row = tile * 16 + (lane & 15);          // n within chunk
    int kb  = (lane >> 4) * 16;
    const _Float16* p = hs + row * HST + kstep * 32 + kb;
    v16h bm;
    #pragma unroll
    for (int v = 0; v < 8; ++v) { bm[2 * v] = p[2 * v]; bm[2 * v + 1] = p[2 * v + 1]; }
    return bm;
}

// ---------------------------------------------------------------------------
// Chunk staging: 32 rows x 384 halfs = 1536 16B vectors; 128 threads x 12.
// Async path issues GLOBAL_LOAD_ASYNC_TO_LDS_B128 (ASYNCcnt); caller overlaps
// WMMA compute on the other buffer, then waits asynccnt==0 before the barrier.
// ---------------------------------------------------------------------------
__device__ inline void stage_chunk_issue(_Float16* __restrict__ hs,
                                         const _Float16* __restrict__ hb,
                                         int c, int tid) {
    const _Float16* gsrc = hb + (long)(c * 32) * D_;
#if USE_ASYNC_LDS
    #pragma unroll
    for (int it = 0; it < 12; ++it) {
        int vi = tid + it * 128;
        int r = vi / 48, c8 = vi % 48;                 // 48 uint4 per 384-half row
        AS1 i32x4* g =
            (AS1 i32x4*)(unsigned long long)(gsrc + r * D_ + c8 * 8);
        AS3 i32x4* l =
            (AS3 i32x4*)(unsigned int)(unsigned long long)(hs + r * HST + c8 * 8);
        __builtin_amdgcn_global_load_async_to_lds_b128(g, l, 0, 0);
    }
#else
    const uint4* src = (const uint4*)gsrc;
    #pragma unroll
    for (int it = 0; it < 12; ++it) {
        int vi = tid + it * 128;
        int r = vi / 48, c8 = vi % 48;
        *(uint4*)(hs + r * HST + c8 * 8) = src[r * 48 + c8];
    }
#endif
}

__device__ inline void stage_chunk_wait() {
#if USE_ASYNC_LDS
    __builtin_amdgcn_s_wait_asynccnt(0);
#endif
}

// ---------------------------------------------------------------------------
// Pass 1: streaming logsumexp of scores = U_w . h^T over n, per (b, y) row.
// Grid (B_, YT_), 128 threads = 4 waves; wave owns 16 y-rows. Double-buffered
// LDS chunks: async-load chunk c+1 while WMMA-consuming chunk c.
// ---------------------------------------------------------------------------
__global__ void __launch_bounds__(128)
attn_pass1(const _Float16* __restrict__ h, const float* __restrict__ Uw,
           float* __restrict__ rowmax, float* __restrict__ rowsum) {
    __shared__ _Float16 hs[2][32 * HST];
    int b = blockIdx.x, yt = blockIdx.y;
    int tid = threadIdx.x, lane = tid & 31, wave = tid >> 5;
    int y0 = yt * 64 + wave * 16;

    v16h ua[KSTEPS];                       // persistent U_w A-fragments
    #pragma unroll
    for (int ks = 0; ks < KSTEPS; ++ks) ua[ks] = load_a_frag(Uw, y0, lane, ks, Y_);

    float M[8], S[8];
    #pragma unroll
    for (int v = 0; v < 8; ++v) { M[v] = -1e30f; S[v] = 0.f; }

    const _Float16* hb = h + (long)b * NP_ * D_;
    stage_chunk_issue(hs[0], hb, 0, tid);
    stage_chunk_wait();
    __syncthreads();

    for (int c = 0; c < NCHUNK; ++c) {
        const _Float16* cur = hs[c & 1];
        if (c + 1 < NCHUNK) stage_chunk_issue(hs[(c + 1) & 1], hb, c + 1, tid);

        v8f a0 = {}, a1 = {};
        #pragma unroll
        for (int ks = 0; ks < KSTEPS; ++ks) {
            v16h bm0 = load_b_frag(cur, 0, lane, ks);
            a0 = __builtin_amdgcn_wmma_f32_16x16x32_f16(false, ua[ks], false, bm0,
                                                        (short)0, a0, false, false);
            v16h bm1 = load_b_frag(cur, 1, lane, ks);
            a1 = __builtin_amdgcn_wmma_f32_16x16x32_f16(false, ua[ks], false, bm1,
                                                        (short)0, a1, false, false);
        }
        int n0 = c * 32 + (lane & 15);
        int n1 = n0 + 16;
        #pragma unroll
        for (int v = 0; v < 8; ++v) {      // per-lane streaming (M, S); merge at end
            float s0 = (n0 < N_) ? a0[v] : -1e30f;
            float s1 = (n1 < N_) ? a1[v] : -1e30f;
            float mn = fmaxf(M[v], fmaxf(s0, s1));
            S[v] = S[v] * __expf(M[v] - mn) + __expf(s0 - mn) + __expf(s1 - mn);
            M[v] = mn;
        }
        stage_chunk_wait();
        __syncthreads();
    }
    // merge the 16 lane-local (M,S) pairs of each half-wave
    #pragma unroll
    for (int v = 0; v < 8; ++v) {
        float m = M[v], s = S[v];
        #pragma unroll
        for (int msk = 1; msk <= 8; msk <<= 1) {
            float mo = __shfl_xor(m, msk, 32);
            float so = __shfl_xor(s, msk, 32);
            float mn = fmaxf(m, mo);
            s = s * __expf(m - mn) + so * __expf(mo - mn);
            m = mn;
        }
        if ((lane & 15) == 0) {
            int y = y0 + v + 8 * (lane >> 4);
            if (y < Y_) { rowmax[(long)b * Y_ + y] = m; rowsum[(long)b * Y_ + y] = s; }
        }
    }
}

// ---------------------------------------------------------------------------
// Pass 2: y[b,y] = sum_n softmax(s)[n] * (final_w[y] . h[b,n]) + final_b[y].
// Two WMMA GEMMs (U_w and final_w) share each B fragment. Same double buffer.
// ---------------------------------------------------------------------------
__global__ void __launch_bounds__(128)
attn_pass2(const _Float16* __restrict__ h, const float* __restrict__ Uw,
           const float* __restrict__ Fw, const float* __restrict__ fbias,
           const float* __restrict__ rowmax, const float* __restrict__ rowsum,
           float* __restrict__ out) {
    __shared__ _Float16 hs[2][32 * HST];
    int b = blockIdx.x, yt = blockIdx.y;
    int tid = threadIdx.x, lane = tid & 31, wave = tid >> 5;
    int y0 = yt * 64 + wave * 16;

    v16h ua[KSTEPS], wa[KSTEPS];
    #pragma unroll
    for (int ks = 0; ks < KSTEPS; ++ks) {
        ua[ks] = load_a_frag(Uw, y0, lane, ks, Y_);
        wa[ks] = load_a_frag(Fw, y0, lane, ks, Y_);
    }

    float Mv[8], Ri[8], yp[8];
    #pragma unroll
    for (int v = 0; v < 8; ++v) {
        int y = y0 + v + 8 * (lane >> 4);
        if (y < Y_) {
            Mv[v] = rowmax[(long)b * Y_ + y];
            float s = rowsum[(long)b * Y_ + y];
            Ri[v] = (s > 0.f) ? 1.f / s : 0.f;
        } else { Mv[v] = 0.f; Ri[v] = 0.f; }
        yp[v] = 0.f;
    }

    const _Float16* hb = h + (long)b * NP_ * D_;
    stage_chunk_issue(hs[0], hb, 0, tid);
    stage_chunk_wait();
    __syncthreads();

    for (int c = 0; c < NCHUNK; ++c) {
        const _Float16* cur = hs[c & 1];
        if (c + 1 < NCHUNK) stage_chunk_issue(hs[(c + 1) & 1], hb, c + 1, tid);

        v8f a0 = {}, a1 = {}, g0 = {}, g1 = {};
        #pragma unroll
        for (int ks = 0; ks < KSTEPS; ++ks) {
            v16h bm0 = load_b_frag(cur, 0, lane, ks);
            a0 = __builtin_amdgcn_wmma_f32_16x16x32_f16(false, ua[ks], false, bm0,
                                                        (short)0, a0, false, false);
            g0 = __builtin_amdgcn_wmma_f32_16x16x32_f16(false, wa[ks], false, bm0,
                                                        (short)0, g0, false, false);
            v16h bm1 = load_b_frag(cur, 1, lane, ks);
            a1 = __builtin_amdgcn_wmma_f32_16x16x32_f16(false, ua[ks], false, bm1,
                                                        (short)0, a1, false, false);
            g1 = __builtin_amdgcn_wmma_f32_16x16x32_f16(false, wa[ks], false, bm1,
                                                        (short)0, g1, false, false);
        }
        int n0 = c * 32 + (lane & 15);
        int n1 = n0 + 16;
        #pragma unroll
        for (int v = 0; v < 8; ++v) {
            float al0 = (n0 < N_) ? __expf(a0[v] - Mv[v]) * Ri[v] : 0.f;
            float al1 = (n1 < N_) ? __expf(a1[v] - Mv[v]) * Ri[v] : 0.f;
            yp[v] += al0 * g0[v] + al1 * g1[v];
        }
        stage_chunk_wait();
        __syncthreads();
    }
    #pragma unroll
    for (int v = 0; v < 8; ++v) {
        float s = yp[v];
        #pragma unroll
        for (int msk = 1; msk <= 8; msk <<= 1) s += __shfl_xor(s, msk, 32);
        if ((lane & 15) == 0) {
            int y = y0 + v + 8 * (lane >> 4);
            if (y < Y_) out[(long)b * Y_ + y] = s + fbias[y];
        }
    }
}

// ---------------------------------------------------------------------------
// Kernel 4: BCE-with-logits mean loss over the B*Y logits -> d_out[B*Y]
// ---------------------------------------------------------------------------
__global__ void bce_loss(const float* __restrict__ logits,
                         const float* __restrict__ tgt,
                         float* __restrict__ out) {
    __shared__ float red[256];
    const int total = B_ * Y_;
    float acc = 0.f;
    for (int i = threadIdx.x; i < total; i += 256) {
        float v = logits[i], t = tgt[i];
        acc += fmaxf(v, 0.f) - v * t + log1pf(__expf(-fabsf(v)));
    }
    red[threadIdx.x] = acc;
    __syncthreads();
    for (int s = 128; s > 0; s >>= 1) {
        if ((int)threadIdx.x < s) red[threadIdx.x] += red[threadIdx.x + s];
        __syncthreads();
    }
    if (threadIdx.x == 0) out[total] = red[0] / (float)total;
}

// ---------------------------------------------------------------------------
extern "C" void kernel_launch(void* const* d_in, const int* in_sizes, int n_in,
                              void* d_out, int out_size, void* d_ws, size_t ws_size,
                              hipStream_t stream) {
    (void)in_sizes; (void)n_in; (void)out_size; (void)ws_size;
    // setup_inputs() dict order:
    const int*   x    = (const int*)  d_in[0];
    const float* tgt  = (const float*)d_in[1];
    const float* Wemb = (const float*)d_in[2];
    const float* Uw   = (const float*)d_in[3];
    const float* Fw   = (const float*)d_in[4];
    const float* fb   = (const float*)d_in[5];
    const float* w3   = (const float*)d_in[6];
    const float* b3   = (const float*)d_in[7];
    const float* w5   = (const float*)d_in[8];
    const float* b5   = (const float*)d_in[9];
    const float* w7   = (const float*)d_in[10];
    const float* b7   = (const float*)d_in[11];
    float* out = (float*)d_out;

    // workspace carve: h (f16, 15.5 MB) | rowmax | rowsum
    char* ws = (char*)d_ws;
    _Float16* h = (_Float16*)ws;
    size_t hbytes = (size_t)B_ * NP_ * D_ * sizeof(_Float16);
    float* rowmax = (float*)(ws + hbytes);
    float* rowsum = rowmax + (size_t)B_ * Y_;

    long total = (long)B_ * NP_ * D_;
    conv_embed_tanh<<<dim3((unsigned)((total + 255) / 256)), dim3(256), 0, stream>>>(
        x, Wemb, w3, b3, w5, b5, w7, b7, h);

    dim3 grid(B_, YT_);
    attn_pass1<<<grid, dim3(128), 0, stream>>>(h, Uw, rowmax, rowsum);
    attn_pass2<<<grid, dim3(128), 0, stream>>>(h, Uw, Fw, fb, rowmax, rowsum, out);
    bce_loss<<<1, 256, 0, stream>>>(out, tgt, out);
}
